// GraphLearningModule_SelfAtt_Kernel_32512902431527
// MI455X (gfx1250) — matrix operand order repaired and checked by reading, not compiled, not run on hardware
//
#include <hip/hip_runtime.h>

typedef __attribute__((ext_vector_type(16))) __bf16 v16bf;
typedef __attribute__((ext_vector_type(8)))  float  v8f;

#define BATCH 8
#define NNODE 4096
#define TDIM  128
#define DDIM  32
// RBF_GAMMA * log2(e): fold gamma and the exp->exp2 conversion into one constant
#define GAMMA_LOG2E 0.00144269504088896f
#define DIAG_VALUE 0.1f
#define LDS_STRIDE 20   // dwords per 32-bf16 row (padded from 16: bank-spread, 16B aligned)

// ---------------------------------------------------------------------------
// Kernel 1: fused projection  p = x @ W + b  and row squared-norms.
// One wave (32 lanes = 32 latent dims) per (row, projection) pair.
// ---------------------------------------------------------------------------
__global__ void __launch_bounds__(256)
proj_kernel(const float* __restrict__ x,
            const float* __restrict__ W1, const float* __restrict__ b1,
            const float* __restrict__ W2, const float* __restrict__ b2,
            float* __restrict__ p1, float* __restrict__ p2,
            float* __restrict__ sq1, float* __restrict__ sq2)
{
    const int tid  = threadIdx.x;
    const int wave = tid >> 5;
    const int lane = tid & 31;
    const int row   = blockIdx.x * 4 + (wave >> 1);   // 0 .. B*N-1
    const int which = wave & 1;                       // 0 -> W1/p1, 1 -> W2/p2

    const float* __restrict__ W    = which ? W2 : W1;
    const float* __restrict__ bias = which ? b2 : b1;
    const float* __restrict__ xr   = x + (size_t)row * TDIM;

    float acc = bias[lane];
#pragma unroll 8
    for (int k = 0; k < TDIM; ++k)
        acc = fmaf(xr[k], W[k * DDIM + lane], acc);

    float* __restrict__ p = which ? p2 : p1;
    p[(size_t)row * DDIM + lane] = acc;

    float s = acc * acc;
#pragma unroll
    for (int off = 16; off > 0; off >>= 1)
        s += __shfl_xor(s, off, 32);
    if (lane == 0)
        (which ? sq2 : sq1)[row] = s;
}

// f32 -> packed bf16 pair (round-to-nearest-even)
__device__ __forceinline__ unsigned int pack_bf16(float lo, float hi)
{
    unsigned int a = __float_as_uint(lo);
    unsigned int b = __float_as_uint(hi);
    a = (a + 0x7FFFu + ((a >> 16) & 1u)) >> 16;
    b = (b + 0x7FFFu + ((b >> 16) & 1u)) >> 16;
    return a | (b << 16);
}

// ---------------------------------------------------------------------------
// Kernel 2: block = 128x128 output macro-tile, 8 waves of 16 rows each.
// Per batch: stage A/B panels (128x32) into LDS as bf16 (converted once),
// then each wave issues 8 x v_wmma_f32_16x16x32_bf16 from ds_load_b128 frags.
// B fragments are double-buffered (prefetch j+1 while WMMA j runs) so the
// LDS latency overlaps the WMMA + exp epilogue instead of a full dscnt wait.
// ---------------------------------------------------------------------------
__global__ void __launch_bounds__(256)
rbf_kernel(const float* __restrict__ p1, const float* __restrict__ p2,
           const float* __restrict__ sq1, const float* __restrict__ sq2,
           float* __restrict__ out)
{
    __shared__ unsigned int ldsA[128 * LDS_STRIDE];   // 10 KB
    __shared__ unsigned int ldsB[128 * LDS_STRIDE];   // 10 KB

    const int tid  = threadIdx.x;
    const int wave = tid >> 5;
    const int lane = tid & 31;
    const int hi   = lane >> 4;      // lane half select
    const int m    = lane & 15;

    const int R0 = (blockIdx.x >> 5) << 7;   // macro-tile row base (32x32 grid)
    const int C0 = (blockIdx.x & 31) << 7;   // macro-tile col base

    // staging role: each thread converts half of one 32-float row (64B load)
    const int srow  = tid >> 1;
    const int shalf = tid & 1;

    float res[8][8];
#pragma unroll
    for (int j = 0; j < 8; ++j)
#pragma unroll
        for (int r = 0; r < 8; ++r) res[j][r] = 0.0f;

    union Frag { uint4 q[2]; v16bf v; };

    for (int b = 0; b < BATCH; ++b) {
        if (b) __syncthreads();   // previous iteration's LDS reads done

        // ---- stage A panel: p1[b, R0..R0+127, :] -> bf16 in LDS ----
        {
            const float4* s = (const float4*)(p1 +
                (((size_t)b * NNODE + R0 + srow) << 5) + shalf * 16);
            float4 f0 = s[0], f1 = s[1], f2 = s[2], f3 = s[3];
            uint4 u0, u1;
            u0.x = pack_bf16(f0.x, f0.y); u0.y = pack_bf16(f0.z, f0.w);
            u0.z = pack_bf16(f1.x, f1.y); u0.w = pack_bf16(f1.z, f1.w);
            u1.x = pack_bf16(f2.x, f2.y); u1.y = pack_bf16(f2.z, f2.w);
            u1.z = pack_bf16(f3.x, f3.y); u1.w = pack_bf16(f3.z, f3.w);
            uint4* d = (uint4*)&ldsA[srow * LDS_STRIDE + shalf * 8];
            d[0] = u0; d[1] = u1;
        }
        // ---- stage B panel: p2[b, C0..C0+127, :] -> bf16 in LDS ----
        {
            const float4* s = (const float4*)(p2 +
                (((size_t)b * NNODE + C0 + srow) << 5) + shalf * 16);
            float4 f0 = s[0], f1 = s[1], f2 = s[2], f3 = s[3];
            uint4 u0, u1;
            u0.x = pack_bf16(f0.x, f0.y); u0.y = pack_bf16(f0.z, f0.w);
            u0.z = pack_bf16(f1.x, f1.y); u0.w = pack_bf16(f1.z, f1.w);
            u1.x = pack_bf16(f2.x, f2.y); u1.y = pack_bf16(f2.z, f2.w);
            u1.z = pack_bf16(f3.x, f3.y); u1.w = pack_bf16(f3.z, f3.w);
            uint4* d = (uint4*)&ldsB[srow * LDS_STRIDE + shalf * 8];
            d[0] = u0; d[1] = u1;
        }
        __syncthreads();

        // ---- A fragment for this wave's 16 rows (ISA 16-bit A 16x32 layout):
        //      lane(m,hi): K = hi*8+[0..7] and 16+hi*8+[0..7]
        Frag Af;
        const unsigned int* arow = &ldsA[(wave * 16 + m) * LDS_STRIDE];
        Af.q[0] = *(const uint4*)(arow + hi * 4);
        Af.q[1] = *(const uint4*)(arow + 8 + hi * 4);

        // row norms, pre-scaled by -gamma*log2(e)
        const float* __restrict__ s1p =
            sq1 + (size_t)b * NNODE + R0 + wave * 16 + hi * 8;
        float4 s1a = *(const float4*)(s1p);
        float4 s1b = *(const float4*)(s1p + 4);
        float g1[8] = { s1a.x, s1a.y, s1a.z, s1a.w,
                        s1b.x, s1b.y, s1b.z, s1b.w };
#pragma unroll
        for (int r = 0; r < 8; ++r) g1[r] *= -GAMMA_LOG2E;

        // B fragment (32x16): lane(n=m,hi): K = hi*8+[0..7] and 16+hi*8+[0..7].
        // Double-buffered: prefetch j+1 before consuming j.
        const unsigned int* bbase = &ldsB[m * LDS_STRIDE + hi * 4];
        Frag Bf[2];
        Bf[0].q[0] = *(const uint4*)(bbase);
        Bf[0].q[1] = *(const uint4*)(bbase + 8);

#pragma unroll
        for (int j = 0; j < 8; ++j) {
            if (j < 7) {
                const unsigned int* bnxt = bbase + (j + 1) * 16 * LDS_STRIDE;
                Bf[(j + 1) & 1].q[0] = *(const uint4*)(bnxt);
                Bf[(j + 1) & 1].q[1] = *(const uint4*)(bnxt + 8);
            }

            v8f cz = {};
            v8f dot = __builtin_amdgcn_wmma_f32_16x16x32_bf16(
                false, Af.v, false, Bf[j & 1].v, (short)0, cz, false, false);

            const float g2 =
                -GAMMA_LOG2E * sq2[(size_t)b * NNODE + C0 + j * 16 + m];
#pragma unroll
            for (int r = 0; r < 8; ++r) {
                // t = -G*(s1+s2) + 2G*dot  ==  -G * dist^2 (before clamp)
                float t = fmaf(2.0f * GAMMA_LOG2E, dot[r], g1[r] + g2);
                t = fminf(t, 0.0f);                    // == -G*max(dist^2, 0)
                res[j][r] += __builtin_amdgcn_exp2f(t); // raw v_exp_f32
            }
        }
    }

    // ---- epilogue: batch mean, diagonal fill, coalesced stores ----
    const int col_base = C0 + m;
#pragma unroll
    for (int j = 0; j < 8; ++j) {
        const int col = col_base + j * 16;
#pragma unroll
        for (int r = 0; r < 8; ++r) {
            const int row = R0 + wave * 16 + hi * 8 + r;
            float v = res[j][r] * (1.0f / (float)BATCH);
            if (row == col) v = DIAG_VALUE;
            out[(size_t)row * NNODE + col] = v;
        }
    }
}

extern "C" void kernel_launch(void* const* d_in, const int* in_sizes, int n_in,
                              void* d_out, int out_size, void* d_ws, size_t ws_size,
                              hipStream_t stream)
{
    const float* x  = (const float*)d_in[0];
    const float* W1 = (const float*)d_in[1];
    const float* b1 = (const float*)d_in[2];
    const float* W2 = (const float*)d_in[3];
    const float* b2 = (const float*)d_in[4];
    float* out = (float*)d_out;

    // workspace layout (fp32):  p1 | p2 | sq1 | sq2   (~8.65 MB total)
    float* ws  = (float*)d_ws;
    const size_t pElems = (size_t)BATCH * NNODE * DDIM;   // 1,048,576
    float* p1  = ws;
    float* p2  = p1 + pElems;
    float* sq1 = p2 + pElems;
    float* sq2 = sq1 + (size_t)BATCH * NNODE;

    // Phase 1: projections. 32768 rows * 2 projections, 1 wave each.
    proj_kernel<<<(BATCH * NNODE) / 4, 256, 0, stream>>>(
        x, W1, b1, W2, b2, p1, p2, sq1, sq2);

    // Phase 2: 32x32 macro-tiles of 128x128, 8 waves/block.
    rbf_kernel<<<32 * 32, 256, 0, stream>>>(p1, p2, sq1, sq2, out);
}
